// MultiHeadSelfAttention_8254927143124
// MI455X (gfx1250) — compile-verified, matrix operand order checked by reading, not run on hardware
//
#include <hip/hip_runtime.h>
#include <hip/hip_bf16.h>
#include <stdint.h>

// MHA: B=2, S=2048, E=1024, H=16, Dh=64.  All matmuls on v_wmma_f32_16x16x32_bf16,
// f32 accumulate, flash-attention so S x S scores never hit HBM.
// Tile staging uses gfx1250 GLOBAL_LOAD_ASYNC_TO_LDS_B128 (ASYNCcnt) with
// double-buffered LDS in the GEMM so copies overlap the WMMA stream.

typedef __attribute__((ext_vector_type(16))) __bf16 v16bf;
typedef __attribute__((ext_vector_type(8)))  float  v8f;

#define EMBED 1024
#define SEQ   2048
#define NHEAD 16
#define HDIM  64

static __device__ __forceinline__ v8f wmma_bf16(v16bf a, v16bf b, v8f c) {
  // 8 args: (neg_a, A, neg_b, B, c_mod, C, reuse_a, reuse_b)
  return __builtin_amdgcn_wmma_f32_16x16x32_bf16(false, a, false, b, (short)0, c,
                                                 false, false);
}

// Per ISA: generic LDS address low 32 bits == LDS byte address.
static __device__ __forceinline__ uint32_t lds_addr(const void* p) {
  return (uint32_t)(uintptr_t)p;
}

// Async copy of one 16B chunk: global -> LDS, tracked by ASYNCcnt.
static __device__ __forceinline__ void async_b128(uint32_t lds, const void* gp) {
  asm volatile("global_load_async_to_lds_b128 %0, %1, off"
               :: "v"(lds), "v"(gp) : "memory");
}
static __device__ __forceinline__ void wait_async0() {
  asm volatile("s_wait_asynccnt 0x0" ::: "memory");
}

// Copy a 64-row x 32-col bf16 tile (row stride 'gstride' elems) into LDS with
// row stride 40 (padded, conflict-free).  256 x 16B chunks / 128 threads.
static __device__ __forceinline__ void async_tile_64x32(
    const __bf16* g, int gstride, __bf16* lbase, int tid) {
#pragma unroll
  for (int it = 0; it < 2; ++it) {
    int i = tid + 128 * it;
    int row = i >> 2;
    int c8 = (i & 3) * 8;
    async_b128(lds_addr(lbase + row * 40 + c8),
               g + (size_t)row * gstride + c8);
  }
}

// Copy a 64-row x 64-col bf16 tile (contiguous rows of 64) into LDS with
// row stride 72 (padded).  512 x 16B chunks / 128 threads.
static __device__ __forceinline__ void async_tile_64x64(
    const __bf16* g, __bf16* lbase, int tid) {
#pragma unroll
  for (int it = 0; it < 4; ++it) {
    int i = tid + 128 * it;
    int row = i >> 3;
    int c8 = (i & 7) * 8;
    async_b128(lds_addr(lbase + row * 72 + c8),
               g + (size_t)row * 64 + c8);
  }
}

// Load a 16x32 bf16 fragment (A-layout / mirrored B-layout) from LDS.
// Lane L -> row (L%16), K-half (L>>4); dwords at k-offsets {kh*8+0,2,4,6}
// and {16+kh*8+0,2,4,6}.
static __device__ __forceinline__ v16bf load_frag(const __bf16* base, int row,
                                                  int stride, int koff) {
  const int lane = threadIdx.x & 31;
  const __bf16* p = base + (size_t)(row + (lane & 15)) * stride + koff +
                    ((lane >> 4) & 1) * 8;
  union { v16bf v; uint32_t u[8]; } f;
  const uint32_t* pu = (const uint32_t*)p;
#pragma unroll
  for (int i = 0; i < 4; ++i) f.u[i] = pu[i];
  pu = (const uint32_t*)(p + 16);
#pragma unroll
  for (int i = 0; i < 4; ++i) f.u[4 + i] = pu[i];
  return f.v;
}

__global__ void cvt_f32_bf16(const float* __restrict__ s, __bf16* __restrict__ d,
                             int n) {
  int i = blockIdx.x * blockDim.x + threadIdx.x;
  if (i < n) d[i] = (__bf16)s[i];
}

// d[c*R + r] = (bf16)s[r*C + c]  (weights -> W^T so GEMM B-tiles are
// K-contiguous and can be async-copied without an LDS transpose)
__global__ void cvt_transpose_f32_bf16(const float* __restrict__ s,
                                       __bf16* __restrict__ d, int R, int C) {
  int idx = blockIdx.x * blockDim.x + threadIdx.x;
  if (idx < R * C) {
    int r = idx / C, c = idx % C;
    d[(size_t)c * R + r] = (__bf16)s[idx];
  }
}

// C[M,N] = A[M,K] @ W^T[N,K]^T + bias.  64x64 tile per block, 128 threads
// (4 waves), each wave a 32x32 quadrant (2x2 WMMA accs), K-step 32,
// double-buffered async tiles.
// MODE 0: write bf16, head-major [B,H,S,Dh].  MODE 1: write f32 row-major.
template <int MODE>
__global__ __launch_bounds__(128) void gemm_bias_kernel(
    const __bf16* __restrict__ A, const __bf16* __restrict__ Wt,
    const float* __restrict__ bias, void* __restrict__ out, int M, int N, int K) {
  __shared__ __bf16 As[2][64 * 40];
  __shared__ __bf16 Bt[2][64 * 40];

  const int tid  = threadIdx.x;
  const int lane = tid & 31;
  const int wave = tid >> 5;
  const int mbase = blockIdx.y * 64;
  const int nbase = blockIdx.x * 64;
  const int wm = (wave & 1) * 32;
  const int wn = (wave >> 1) * 32;

  v8f acc00 = {}, acc01 = {}, acc10 = {}, acc11 = {};

  const __bf16* Ab = A + (size_t)mbase * K;
  const __bf16* Wb = Wt + (size_t)nbase * K;

  async_tile_64x32(Ab, K, As[0], tid);
  async_tile_64x32(Wb, K, Bt[0], tid);
  wait_async0();
  __syncthreads();

  int cur = 0;
  for (int kb = 0; kb < K; kb += 32) {
    if (kb + 32 < K) {  // overlap next tile's DMA with this tile's WMMAs
      async_tile_64x32(Ab + kb + 32, K, As[cur ^ 1], tid);
      async_tile_64x32(Wb + kb + 32, K, Bt[cur ^ 1], tid);
    }
    v16bf a0 = load_frag(As[cur], wm, 40, 0);
    v16bf a1 = load_frag(As[cur], wm + 16, 40, 0);
    v16bf b0 = load_frag(Bt[cur], wn, 40, 0);
    v16bf b1 = load_frag(Bt[cur], wn + 16, 40, 0);
    acc00 = wmma_bf16(a0, b0, acc00);
    acc01 = wmma_bf16(a0, b1, acc01);
    acc10 = wmma_bf16(a1, b0, acc10);
    acc11 = wmma_bf16(a1, b1, acc11);
    wait_async0();
    __syncthreads();
    cur ^= 1;
  }

  // Epilogue. C-layout: lane L -> col (L&15), rows r + 8*(L>>4), r=0..7.
  const int nl = lane & 15;
  const int mhalf = (lane >> 4) * 8;
  v8f accs[2][2] = {{acc00, acc01}, {acc10, acc11}};
#pragma unroll
  for (int mi = 0; mi < 2; ++mi)
#pragma unroll
    for (int ni = 0; ni < 2; ++ni)
#pragma unroll
      for (int r = 0; r < 8; ++r) {
        int mg = mbase + wm + mi * 16 + r + mhalf;
        int ng = nbase + wn + ni * 16 + nl;
        float v = accs[mi][ni][r] + bias[ng];
        if (MODE == 0) {
          int bb = mg >> 11, ss = mg & (SEQ - 1);
          int hh = ng >> 6,  dd = ng & (HDIM - 1);
          ((__bf16*)out)[((size_t)(bb * NHEAD + hh) * SEQ + ss) * HDIM + dd] =
              (__bf16)v;
        } else {
          ((float*)out)[(size_t)mg * N + ng] = v;
        }
      }
}

// Flash attention per (b,h): 64 query rows per block, stream 64-key K/V tiles.
// Q and K tiles arrive via async-to-LDS; V is transposed through VGPRs.
__global__ __launch_bounds__(128) void attn_kernel(
    const __bf16* __restrict__ Q, const __bf16* __restrict__ Km,
    const __bf16* __restrict__ V, __bf16* __restrict__ O) {
  __shared__ __bf16 Qs[64 * 72];       // [q][d]
  __shared__ __bf16 Ks[64 * 72];       // [kj][d]  (row-major == B-frag layout)
  __shared__ __bf16 Vt[64 * 72];       // [d][kj]  (transposed for PV B-frags)
  __shared__ __bf16 Pw[4][16 * 72];    // per-wave P: [m][kj]

  const int tid  = threadIdx.x;
  const int lane = tid & 31;
  const int wave = tid >> 5;
  const int bh = blockIdx.x;           // b*NHEAD + h
  const int qbase = blockIdx.y * 64;
  const size_t headoff = (size_t)bh * SEQ * HDIM;
  const __bf16* Qh = Q + headoff;
  const __bf16* Kh = Km + headoff;
  const __bf16* Vh = V + headoff;
  const int m0 = wave * 16;

  // Q tile resident for the whole block (async; waited in first iteration)
  async_tile_64x64(Qh + (size_t)qbase * HDIM, Qs, tid);

  float rmax[8], rsum[8];
#pragma unroll
  for (int r = 0; r < 8; ++r) { rmax[r] = -3.0e38f; rsum[r] = 0.0f; }
  v8f o0 = {}, o1 = {}, o2 = {}, o3 = {};

  for (int kt = 0; kt < SEQ; kt += 64) {
    __syncthreads();
    async_tile_64x64(Kh + (size_t)kt * HDIM, Ks, tid);
#pragma unroll
    for (int it = 0; it < 16; ++it) {
      int i = tid + 128 * it;
      int row = i >> 5;                 // key index within tile
      int d2 = (i & 31) * 2;
      uint32_t vv = *(const uint32_t*)&Vh[(size_t)(kt + row) * HDIM + d2];
      ((uint16_t*)Vt)[d2 * 72 + row]       = (uint16_t)(vv & 0xffffu);
      ((uint16_t*)Vt)[(d2 + 1) * 72 + row] = (uint16_t)(vv >> 16);
    }
    wait_async0();
    __syncthreads();

    // scores: 16 (q) x 64 (k) = 4 accumulators; K dim = Dh = 64 -> 2 steps
    v8f s0 = {}, s1 = {}, s2 = {}, s3 = {};
#pragma unroll
    for (int dstep = 0; dstep < HDIM; dstep += 32) {
      v16bf aq = load_frag(Qs, m0, 72, dstep);
      s0 = wmma_bf16(aq, load_frag(Ks, 0, 72, dstep), s0);
      s1 = wmma_bf16(aq, load_frag(Ks, 16, 72, dstep), s1);
      s2 = wmma_bf16(aq, load_frag(Ks, 32, 72, dstep), s2);
      s3 = wmma_bf16(aq, load_frag(Ks, 48, 72, dstep), s3);
    }
    const float scale = 0.125f;  // 1/sqrt(64)
    s0 *= scale; s1 *= scale; s2 *= scale; s3 *= scale;

    // online softmax; each row lives in 16 lanes of one half -> xor 1,2,4,8
    float corr[8];
#pragma unroll
    for (int r = 0; r < 8; ++r) {
      float lm = fmaxf(fmaxf(s0[r], s1[r]), fmaxf(s2[r], s3[r]));
#pragma unroll
      for (int off = 1; off < 16; off <<= 1)
        lm = fmaxf(lm, __shfl_xor(lm, off, 32));
      float mn = fmaxf(rmax[r], lm);
      float c = __expf(rmax[r] - mn);
      float p0 = __expf(s0[r] - mn);
      float p1 = __expf(s1[r] - mn);
      float p2 = __expf(s2[r] - mn);
      float p3 = __expf(s3[r] - mn);
      float ls = p0 + p1 + p2 + p3;
#pragma unroll
      for (int off = 1; off < 16; off <<= 1)
        ls += __shfl_xor(ls, off, 32);
      rsum[r] = rsum[r] * c + ls;
      rmax[r] = mn;
      corr[r] = c;
      // C-layout -> A-layout transpose via per-wave LDS
      int m = r + ((lane >> 4) * 8);
      int kj = lane & 15;
      __bf16* pp = &Pw[wave][m * 72];
      pp[kj]      = (__bf16)p0;
      pp[kj + 16] = (__bf16)p1;
      pp[kj + 32] = (__bf16)p2;
      pp[kj + 48] = (__bf16)p3;
    }
#pragma unroll
    for (int r = 0; r < 8; ++r) {
      o0[r] *= corr[r]; o1[r] *= corr[r]; o2[r] *= corr[r]; o3[r] *= corr[r];
    }
    __syncthreads();

    // O += P @ V : A = P[16 x 64(kj)], B = Vt rows d contiguous in kj
#pragma unroll
    for (int dstep = 0; dstep < 64; dstep += 32) {
      v16bf ap = load_frag(Pw[wave], 0, 72, dstep);
      o0 = wmma_bf16(ap, load_frag(Vt, 0, 72, dstep), o0);
      o1 = wmma_bf16(ap, load_frag(Vt, 16, 72, dstep), o1);
      o2 = wmma_bf16(ap, load_frag(Vt, 32, 72, dstep), o2);
      o3 = wmma_bf16(ap, load_frag(Vt, 48, 72, dstep), o3);
    }
  }

  // normalize + write back merged-head layout [B*S][E] (bf16 for final GEMM)
  const int b = bh >> 4, h = bh & (NHEAD - 1);
  const int nl = lane & 15;
  const int mhalf = (lane >> 4) * 8;
  v8f oo[4] = {o0, o1, o2, o3};
#pragma unroll
  for (int dsub = 0; dsub < 4; ++dsub)
#pragma unroll
    for (int r = 0; r < 8; ++r) {
      int s = qbase + m0 + r + mhalf;
      int d = dsub * 16 + nl;
      float val = oo[dsub][r] / rsum[r];
      O[((size_t)(b * SEQ + s)) * EMBED + h * HDIM + d] = (__bf16)val;
    }
}

extern "C" void kernel_launch(void* const* d_in, const int* in_sizes, int n_in,
                              void* d_out, int out_size, void* d_ws, size_t ws_size,
                              hipStream_t stream) {
  const float* x  = (const float*)d_in[0];
  const float* Wq = (const float*)d_in[1];
  const float* bq = (const float*)d_in[2];
  const float* Wk = (const float*)d_in[3];
  const float* bk = (const float*)d_in[4];
  const float* Wv = (const float*)d_in[5];
  const float* bv = (const float*)d_in[6];
  const float* Wo = (const float*)d_in[7];
  const float* bo = (const float*)d_in[8];

  char* ws = (char*)d_ws;
  const size_t MB = (size_t)1 << 20;
  __bf16* xb  = (__bf16*)(ws + 0);        // 8 MB  : x bf16 [4096,1024]
  __bf16* wqb = (__bf16*)(ws + 8 * MB);   // 2 MB  : Wq^T bf16 [N=1024][K=1024]
  __bf16* wkb = (__bf16*)(ws + 10 * MB);  // 2 MB
  __bf16* wvb = (__bf16*)(ws + 12 * MB);  // 2 MB
  __bf16* wob = (__bf16*)(ws + 14 * MB);  // 2 MB
  __bf16* Qb  = (__bf16*)(ws + 16 * MB);  // 8 MB  : [B,H,S,Dh] bf16
  __bf16* Kb  = (__bf16*)(ws + 24 * MB);  // 8 MB
  __bf16* Vb  = (__bf16*)(ws + 32 * MB);  // 8 MB
  __bf16* AOb = (__bf16*)(ws + 40 * MB);  // 8 MB  : attn out [B*S,E] bf16

  const int NX = 2 * SEQ * EMBED;  // 4,194,304
  const int NW = EMBED * EMBED;    // 1,048,576
  cvt_f32_bf16<<<NX / 256, 256, 0, stream>>>(x, xb, NX);
  cvt_transpose_f32_bf16<<<NW / 256, 256, 0, stream>>>(Wq, wqb, EMBED, EMBED);
  cvt_transpose_f32_bf16<<<NW / 256, 256, 0, stream>>>(Wk, wkb, EMBED, EMBED);
  cvt_transpose_f32_bf16<<<NW / 256, 256, 0, stream>>>(Wv, wvb, EMBED, EMBED);
  cvt_transpose_f32_bf16<<<NW / 256, 256, 0, stream>>>(Wo, wob, EMBED, EMBED);

  dim3 ggrid(EMBED / 64, (2 * SEQ) / 64);  // (16, 64)
  gemm_bias_kernel<0><<<ggrid, 128, 0, stream>>>(xb, wqb, bq, Qb, 2 * SEQ, EMBED, EMBED);
  gemm_bias_kernel<0><<<ggrid, 128, 0, stream>>>(xb, wkb, bk, Kb, 2 * SEQ, EMBED, EMBED);
  gemm_bias_kernel<0><<<ggrid, 128, 0, stream>>>(xb, wvb, bv, Vb, 2 * SEQ, EMBED, EMBED);

  attn_kernel<<<dim3(2 * NHEAD, SEQ / 64), 128, 0, stream>>>(Qb, Kb, Vb, AOb);

  gemm_bias_kernel<1><<<ggrid, 128, 0, stream>>>(AOb, wob, bo, d_out, 2 * SEQ, EMBED, EMBED);
}
